// LSTMModel_88923002896454
// MI455X (gfx1250) — matrix operand order repaired and checked by reading, not compiled, not run on hardware
//
#include <hip/hip_runtime.h>
#include <hip/hip_bf16.h>

typedef _Float16 h8   __attribute__((ext_vector_type(8)));
typedef _Float16 v16h __attribute__((ext_vector_type(16)));
typedef float    v8f  __attribute__((ext_vector_type(8)));

constexpr int Bt   = 512;   // batch
constexpr int Tt   = 512;   // time
constexpr int Hh   = 128;   // hidden
constexpr int G4H  = 512;   // 4*H
constexpr int OUTN = 7;
constexpr int BT   = 32;    // batch tile per workgroup

#define DEVFN __device__ __forceinline__

DEVFN float fsig(float x) { return 1.0f / (1.0f + __expf(-x)); }
DEVFN float ftanh(float x) {
    float e = __expf(-2.0f * x);
    return (1.0f - e) / (1.0f + e);
}

// A fragment (16x32 f16), CDNA5 layout: lane holds row M=lane&15;
// v0..3 = K (lane>=16 ? 8:0)+0..7, v4..7 = K 16+(lane>=16?8:0)+0..7
DEVFN v16h load_a_frag(const _Float16* base, int rowStride, int lane) {
    int m    = lane & 15;
    int hoff = (lane >> 4) * 8;
    const _Float16* p = base + m * rowStride + hoff;
    h8 lo = *(const h8*)(p);
    h8 hi = *(const h8*)(p + 16);
    v16h a;
#pragma unroll
    for (int i = 0; i < 8; ++i) { a[i] = lo[i]; a[i + 8] = hi[i]; }
    return a;
}

// B fragment (32x16 f16): lane holds N=nBase+(lane&15); elements 0..7 = K (lane>=16?8:0)+0..7,
// elements 8..15 = K 16+(lane>=16?8:0)+0..7.
// wT is weights stored K-major: wT[k*512 + n].
DEVFN v16h load_b_frag(const _Float16* wT, int kBase, int nBase, int lane) {
    const _Float16* p = wT + (kBase + (lane >> 4) * 8) * G4H + nBase + (lane & 15);
    v16h b;
#pragma unroll
    for (int i = 0; i < 8; ++i) { b[i] = p[i * G4H]; b[i + 8] = p[(16 + i) * G4H]; }
    return b;
}

// One LSTM layer, fused input projection + recurrence.
// KI = padded input feature K (32 for layer0 [F=8 zero-padded], 128 otherwise).
// Grid: B/BT workgroups x 256 threads (8 waves). Each WG owns a 32-row batch
// tile through all T steps; weights resident in LDS (K-major f16).
template <int KI, bool L0>
__global__ void __launch_bounds__(256, 1)
lstm_layer_kernel(const float* __restrict__ xf32,
                  const _Float16* __restrict__ xf16,
                  const float* __restrict__ wih,
                  const float* __restrict__ whh,
                  const float* __restrict__ bih,
                  const float* __restrict__ bhh,
                  _Float16* __restrict__ out) {
    extern __shared__ _Float16 smem[];
    _Float16* whhT = smem;                 // [128][512] K-major
    _Float16* wihT = whhT + Hh * G4H;      // [KI][512]  K-major
    _Float16* hbuf = wihT + KI * G4H;      // [32][128]  h state
    _Float16* xbuf = hbuf + BT * Hh;       // [32][KI]   x_t staging

    const int tid   = threadIdx.x;
    const int lane  = tid & 31;
    const int w     = tid >> 5;            // wave 0..7 -> hidden col tile
    const int bBase = blockIdx.x * BT;

    // ---- preload W_hh transposed to K-major f16 ----
    for (int idx = tid; idx < G4H * Hh; idx += 256) {
        int n = idx / Hh, k = idx % Hh;    // whh[n][k]
        whhT[k * G4H + n] = (_Float16)whh[idx];
    }
    // ---- preload W_ih (layer0: zero-pad K to 32) ----
    if (L0) {
        for (int idx = tid; idx < KI * G4H; idx += 256) wihT[idx] = (_Float16)0.0f;
        __syncthreads();
        for (int idx = tid; idx < G4H * 8; idx += 256) {
            int n = idx / 8, k = idx % 8;  // wih[n][k], F=8
            wihT[k * G4H + n] = (_Float16)wih[idx];
        }
    } else {
        for (int idx = tid; idx < G4H * KI; idx += 256) {
            int n = idx / KI, k = idx % KI;
            wihT[k * G4H + n] = (_Float16)wih[idx];
        }
    }
    for (int idx = tid; idx < BT * Hh; idx += 256) hbuf[idx] = (_Float16)0.0f;
    if (L0) {
        for (int idx = tid; idx < BT * KI; idx += 256) xbuf[idx] = (_Float16)0.0f;
    }

    // per-lane fused biases (b_ih + b_hh) for this wave's hidden columns;
    // added AFTER the WMMA chain (scalars, minimal register pressure)
    const int nl  = lane & 15;
    const int hc  = w;                 // hidden col tile 0..7
    const int col = hc * 16 + nl;      // hidden column 0..127
    const float b_i = bih[col]           + bhh[col];
    const float b_f = bih[Hh + col]      + bhh[Hh + col];
    const float b_g = bih[2 * Hh + col]  + bhh[2 * Hh + col];
    const float b_o = bih[3 * Hh + col]  + bhh[3 * Hh + col];

    const int nbase[4] = { hc * 16, Hh + hc * 16, 2 * Hh + hc * 16, 3 * Hh + hc * 16 };

    __syncthreads();

    // cell state in registers: [row tile][acc vgpr]
    float cst[2][8];
#pragma unroll
    for (int r = 0; r < 2; ++r)
#pragma unroll
        for (int v = 0; v < 8; ++v) cst[r][v] = 0.0f;

    for (int t = 0; t < Tt; ++t) {
        // ---- stage x_t into LDS ----
        if (L0) {
            if (tid < BT) {
                const float* xr = xf32 + ((size_t)(bBase + tid) * Tt + t) * 8;
                h8 v;
#pragma unroll
                for (int f = 0; f < 8; ++f) v[f] = (_Float16)xr[f];
                *(h8*)(xbuf + tid * KI) = v;   // cols 8..31 stay zero
            }
        } else {
#pragma unroll
            for (int c = 0; c < 2; ++c) {
                int ch  = tid + c * 256;               // 512 chunks of 8 halves
                int row = ch >> 4;
                int off = (ch & 15) * 8;
                const h8* src = (const h8*)(xf16 + (size_t)t * Bt * Hh +
                                            (size_t)(bBase + row) * Hh + off);
                *(h8*)(xbuf + row * KI + off) = *src;
            }
            if (t + 1 < Tt)
                __builtin_prefetch(xf16 + (size_t)(t + 1) * Bt * Hh +
                                   (size_t)bBase * Hh, 0, 1);
        }
        __syncthreads();

        // ---- gates = h @ Whh^T + x @ Wih^T (WMMA f16->f32), zero-init ----
        v8f acc[2][4];
#pragma unroll
        for (int r = 0; r < 2; ++r)
#pragma unroll
            for (int g = 0; g < 4; ++g)
#pragma unroll
                for (int e = 0; e < 8; ++e) acc[r][g][e] = 0.0f;

        // recurrent part: K = 128.  Load ALL fragments of a k-chunk (2 A + 4 B,
        // 4 b128 A loads + 64 strided 16-bit B loads) before the 8 WMMAs so loads overlap the XDL pipe.
#pragma unroll
        for (int kc = 0; kc < Hh / 32; ++kc) {
            v16h a0 = load_a_frag(hbuf + kc * 32,            Hh, lane);
            v16h a1 = load_a_frag(hbuf + 16 * Hh + kc * 32,  Hh, lane);
            v16h bf[4];
#pragma unroll
            for (int g = 0; g < 4; ++g)
                bf[g] = load_b_frag(whhT, kc * 32, nbase[g], lane);
#pragma unroll
            for (int g = 0; g < 4; ++g) {
                acc[0][g] = __builtin_amdgcn_wmma_f32_16x16x32_f16(
                    false, a0, false, bf[g], (short)0, acc[0][g], false, false);
                acc[1][g] = __builtin_amdgcn_wmma_f32_16x16x32_f16(
                    false, a1, false, bf[g], (short)0, acc[1][g], false, false);
            }
        }
        // input part: K = KI
#pragma unroll
        for (int kc = 0; kc < KI / 32; ++kc) {
            v16h a0 = load_a_frag(xbuf + kc * 32,            KI, lane);
            v16h a1 = load_a_frag(xbuf + 16 * KI + kc * 32,  KI, lane);
            v16h bf[4];
#pragma unroll
            for (int g = 0; g < 4; ++g)
                bf[g] = load_b_frag(wihT, kc * 32, nbase[g], lane);
#pragma unroll
            for (int g = 0; g < 4; ++g) {
                acc[0][g] = __builtin_amdgcn_wmma_f32_16x16x32_f16(
                    false, a0, false, bf[g], (short)0, acc[0][g], false, false);
                acc[1][g] = __builtin_amdgcn_wmma_f32_16x16x32_f16(
                    false, a1, false, bf[g], (short)0, acc[1][g], false, false);
            }
        }
        __syncthreads();   // all h reads complete before h overwrite

        // ---- in-register bias + activations + state update, write h ----
#pragma unroll
        for (int r = 0; r < 2; ++r) {
#pragma unroll
            for (int v = 0; v < 8; ++v) {
                float gi = acc[r][0][v] + b_i;
                float gf = acc[r][1][v] + b_f;
                float gg = acc[r][2][v] + b_g;
                float go = acc[r][3][v] + b_o;
                float c  = fsig(gf) * cst[r][v] + fsig(gi) * ftanh(gg);
                cst[r][v] = c;
                float h  = fsig(go) * ftanh(c);
                int m = r * 16 + v + ((lane >> 4) << 3);   // C/D layout row
                hbuf[m * Hh + col] = (_Float16)h;
            }
        }
        __syncthreads();

        // ---- coalesced writeback of h_t ----
#pragma unroll
        for (int c = 0; c < 2; ++c) {
            int ch  = tid + c * 256;
            int row = ch >> 4;
            int off = (ch & 15) * 8;
            *(h8*)(out + (size_t)t * Bt * Hh + (size_t)(bBase + row) * Hh + off) =
                *(const h8*)(hbuf + row * Hh + off);
        }
    }
}

// MLP head on last timestep: y = relu(h @ fc1^T + b1) @ fc2^T + b2
__global__ void head_kernel(const _Float16* __restrict__ hseq,
                            const float* __restrict__ fc1w,
                            const float* __restrict__ fc1b,
                            const float* __restrict__ fc2w,
                            const float* __restrict__ fc2b,
                            float* __restrict__ out) {
    __shared__ float hrow[Hh];
    __shared__ float y1[Hh];
    const int b = blockIdx.x;
    const int j = threadIdx.x;
    hrow[j] = (float)hseq[(size_t)(Tt - 1) * Bt * Hh + (size_t)b * Hh + j];
    __syncthreads();
    float s = fc1b[j];
#pragma unroll 4
    for (int k = 0; k < Hh; ++k) s += hrow[k] * fc1w[j * Hh + k];
    y1[j] = fmaxf(s, 0.0f);
    __syncthreads();
    if (j < OUTN) {
        float s2 = fc2b[j];
#pragma unroll 4
        for (int k = 0; k < Hh; ++k) s2 += y1[k] * fc2w[j * Hh + k];
        out[(size_t)b * OUTN + j] = s2;
    }
}

extern "C" void kernel_launch(void* const* d_in, const int* in_sizes, int n_in,
                              void* d_out, int out_size, void* d_ws, size_t ws_size,
                              hipStream_t stream) {
    const float* x     = (const float*)d_in[0];
    const float* wih0  = (const float*)d_in[1];
    const float* whh0  = (const float*)d_in[2];
    const float* bih0  = (const float*)d_in[3];
    const float* bhh0  = (const float*)d_in[4];
    const float* wih1  = (const float*)d_in[5];
    const float* whh1  = (const float*)d_in[6];
    const float* bih1  = (const float*)d_in[7];
    const float* bhh1  = (const float*)d_in[8];
    const float* wih2  = (const float*)d_in[9];
    const float* whh2  = (const float*)d_in[10];
    const float* bih2  = (const float*)d_in[11];
    const float* bhh2  = (const float*)d_in[12];
    const float* fc1w  = (const float*)d_in[13];
    const float* fc1b  = (const float*)d_in[14];
    const float* fc2w  = (const float*)d_in[15];
    const float* fc2b  = (const float*)d_in[16];

    _Float16* buf0 = (_Float16*)d_ws;                       // [T,B,H] f16 = 64MB
    _Float16* buf1 = buf0 + (size_t)Tt * Bt * Hh;           // ping-pong

    const size_t lds0  = sizeof(_Float16) * (Hh * G4H + 32  * G4H + BT * Hh + BT * 32);
    const size_t lds12 = sizeof(_Float16) * (Hh * G4H + Hh  * G4H + BT * Hh + BT * Hh);

    dim3 grid(Bt / BT);   // 16 workgroups
    dim3 blk(256);        // 8 wave32

    lstm_layer_kernel<32,  true ><<<grid, blk, lds0,  stream>>>(x,       nullptr, wih0, whh0, bih0, bhh0, buf0);
    lstm_layer_kernel<128, false><<<grid, blk, lds12, stream>>>(nullptr, buf0,    wih1, whh1, bih1, bhh1, buf1);
    lstm_layer_kernel<128, false><<<grid, blk, lds12, stream>>>(nullptr, buf1,    wih2, whh2, bih2, bhh2, buf0);
    head_kernel<<<dim3(Bt), dim3(Hh), 0, stream>>>(buf0, fc1w, fc1b, fc2w, fc2b, (float*)d_out);
}
